// PointCloudResNet_15015205667360
// MI455X (gfx1250) — matrix operand order repaired and checked by reading, not compiled, not run on hardware
//
#include <hip/hip_runtime.h>
#include <hip/hip_bf16.h>

// ---------------------------------------------------------------------------
// PointCloudResNet forward for MI455X (gfx1250, wave32, WMMA).
// B=4, N=4096, R=16, G=4096, K=16, H=64, CIN=3, COUT=40.
// All GEMM-shaped work runs through v_wmma_f32_16x16x32_bf16 (f32 accum).
// Weights staged TRANSPOSED in LDS -> B fragments are contiguous ds_load_b128;
// conv input tile is channel-last -> A fragments too. Weight staging is
// amortized: mlp2/neighbor blocks keep weights resident across 8 tiles,
// conv blocks process a 4-z column per (b,y). gelu uses v_exp_f32 + v_rcp_f32
// (no IEEE div sequence). ~13MB working set, L2 resident.
// ---------------------------------------------------------------------------

#define BB   4
#define NN   4096
#define RR   16
#define GG   4096
#define KK   16
#define HH   64
#define CIN  3
#define COUT 40

typedef __bf16 v16bf __attribute__((ext_vector_type(16)));
typedef float  v8f   __attribute__((ext_vector_type(8)));

__device__ __forceinline__ v8f wmma_bf16(v16bf a, v16bf b, v8f c) {
  return __builtin_amdgcn_wmma_f32_16x16x32_bf16(false, a, false, b, (short)0, c,
                                                 false, false);
}

// tanh-gelu in sigmoid form: gelu(x) = x * rcp(1 + exp(-2u)),
// u = 0.79788456*(x + 0.044715 x^3).  v_exp_f32 + v_rcp_f32, branch-free,
// exact at +-inf (exp->inf => rcp->0; exp->0 => rcp(1)=1).
__device__ __forceinline__ float gelu_t(float x) {
  float u2 = 1.5957691216057308f * (x + 0.044715f * x * x * x);
  float s = __builtin_amdgcn_rcpf(1.0f + __expf(-u2));
  return x * s;
}

// A fragment: 16x32 (MxK) from row-major bf16 src[m*lda + k], lda in {40,80}
// so each lane's two 8-element k-runs are 16B-aligned -> ds_load_b128 x2.
// Layout (ISA 7.12.2): lane m = lane&15, half = lane>>4;
// elem e -> k = (e>>3)*16 + half*8 + (e&7).
__device__ __forceinline__ v16bf load_A(const __bf16* src, int lda) {
  int lane = threadIdx.x & 31;
  int m = lane & 15, half = lane >> 4;
  const __bf16* row = src + m * lda + half * 8;
  v16bf a;
#pragma unroll
  for (int e = 0; e < 8; ++e) a[e] = row[e];
#pragma unroll
  for (int e = 0; e < 8; ++e) a[8 + e] = row[16 + e];
  return a;
}

// B fragment: 32x16 (KxN) from TRANSPOSED tile srcT[n*ldt + k].
// Per-lane elem e -> k = (e>>3)*16 + half*8 + (e&7), same as load_A.
__device__ __forceinline__ v16bf load_B_T(const __bf16* srcT, int ldt) {
  int lane = threadIdx.x & 31;
  int n = lane & 15, half = lane >> 4;
  const __bf16* p = srcT + n * ldt + half * 8;
  v16bf b;
#pragma unroll
  for (int e = 0; e < 8; ++e) b[e] = p[e];
#pragma unroll
  for (int e = 0; e < 8; ++e) b[8 + e] = p[16 + e];
  return b;
}

// D fragment element j, lane L -> row = j + 8*(L>>4), col = L&15.

__device__ __forceinline__ float vox_coord(int i) {
  return -1.0f + (2.0f / 15.0f) * (float)i;
}

// ---------------------------------------------------------------------------
// Kernel 1/4: fused 2-layer MLP  Y = gelu(X@W1+b1)@W2+b2
// X f32 [M,Kin] (Kin in {3,64}), output bf16, row-major or [B][64][G] NCDHW.
// Block = 128 thr (4 waves); weights resident, 8 row-tiles per block.
// ---------------------------------------------------------------------------
#define MLP_TILES 8
__global__ void mlp2_kernel(const float* __restrict__ X,
                            const float* __restrict__ W1,
                            const float* __restrict__ B1,
                            const float* __restrict__ W2,
                            const float* __restrict__ B2,
                            __bf16* __restrict__ Y,
                            int M, int Kin, int transposed) {
  __shared__ __align__(16) __bf16 sX[16][80];
  __shared__ __align__(16) __bf16 sW1T[64][80];   // [n][k], k padded w/ zeros
  __shared__ __align__(16) __bf16 sW2T[64][80];
  __shared__ __align__(16) __bf16 sMid[16][80];
  __shared__ float sB1[64], sB2[64];

  int tid = threadIdx.x;
  int Kpad = (Kin <= 32) ? 32 : 64;

  for (int t = tid; t < 64 * 64; t += 128) {
    int k = t >> 6, n = t & 63;
    sW1T[n][k] = (k < Kin) ? (__bf16)W1[k * 64 + n] : (__bf16)0.0f;
    sW2T[n][k] = (__bf16)W2[k * 64 + n];
  }
  if (tid < 64) { sB1[tid] = B1[tid]; sB2[tid] = B2[tid]; }

  int wave = tid >> 5, lane = tid & 31;
  int half = lane >> 4, n16 = lane & 15;
  int nt = wave;

  for (int it = 0; it < MLP_TILES; ++it) {
    int row0 = (blockIdx.x * MLP_TILES + it) * 16;

    for (int t = tid; t < 16 * 64; t += 128) {
      int r = t >> 6, c = t & 63;
      int row = row0 + r;
      float v = (c < Kin && row < M) ? X[row * Kin + c] : 0.0f;
      sX[r][c] = (__bf16)v;
    }
    __syncthreads();

    // layer 1
    v8f acc = {};
    for (int s = 0; s < Kpad / 32; ++s) {
      v16bf a = load_A(&sX[0][s * 32], 80);
      v16bf b = load_B_T(&sW1T[nt * 16][s * 32], 80);
      acc = wmma_bf16(a, b, acc);
    }
#pragma unroll
    for (int j = 0; j < 8; ++j) {
      int m = j + half * 8, col = nt * 16 + n16;
      sMid[m][col] = (__bf16)gelu_t(acc[j] + sB1[col]);
    }
    __syncthreads();

    // layer 2
    v8f acc2 = {};
#pragma unroll
    for (int s = 0; s < 2; ++s) {
      v16bf a = load_A(&sMid[0][s * 32], 80);
      v16bf b = load_B_T(&sW2T[nt * 16][s * 32], 80);
      acc2 = wmma_bf16(a, b, acc2);
    }
#pragma unroll
    for (int j = 0; j < 8; ++j) {
      int m = j + half * 8, col = nt * 16 + n16;
      int row = row0 + m;
      if (row < M) {
        float v = acc2[j] + sB2[col];
        if (transposed) {
          int b = row / GG, g = row - b * GG;
          Y[(size_t)(b * 64 + col) * GG + g] = (__bf16)v;
        } else {
          Y[(size_t)row * 64 + col] = (__bf16)v;
        }
      }
    }
    __syncthreads();
  }
}

// ---------------------------------------------------------------------------
// Kernel 2: KNN. 256 thr/block, one voxel per thread, LDS-tiled points.
// ---------------------------------------------------------------------------
__global__ void knn_kernel(const float* __restrict__ pos,
                           int* __restrict__ idxOut) {
  __shared__ float sPx[256], sPy[256], sPz[256];
  int tid = threadIdx.x;
  int b = blockIdx.x >> 4;
  int g = ((blockIdx.x & 15) << 8) + tid;

  float gx = vox_coord(g >> 8);
  float gy = vox_coord((g >> 4) & 15);
  float gz = vox_coord(g & 15);

  float bd[KK];
  int   bi[KK];
#pragma unroll
  for (int k = 0; k < KK; ++k) { bd[k] = 1e30f; bi[k] = 0; }

  for (int tile = 0; tile < NN / 256; ++tile) {
    int j = tile * 256 + tid;
    const float* pp = pos + ((size_t)b * NN + j) * 3;
    sPx[tid] = pp[0] * 2.0f - 1.0f;
    sPy[tid] = pp[1] * 2.0f - 1.0f;
    sPz[tid] = pp[2] * 2.0f - 1.0f;
    __syncthreads();
    for (int i = 0; i < 256; ++i) {
      float dx = sPx[i] - gx, dy = sPy[i] - gy, dz = sPz[i] - gz;
      float d = dx * dx + dy * dy + dz * dz;
      if (d < bd[KK - 1]) {
        bd[KK - 1] = d; bi[KK - 1] = tile * 256 + i;
#pragma unroll
        for (int k = KK - 1; k > 0; --k) {
          if (bd[k] < bd[k - 1]) {
            float td = bd[k]; bd[k] = bd[k - 1]; bd[k - 1] = td;
            int ti = bi[k]; bi[k] = bi[k - 1]; bi[k - 1] = ti;
          }
        }
      }
    }
    __syncthreads();
  }
#pragma unroll
  for (int k = 0; k < KK; ++k)
    idxOut[((size_t)b * GG + g) * KK + k] = bi[k];
}

// ---------------------------------------------------------------------------
// Kernel 3: fused gather + pe-MLP + (hj*pe)@msg_w + mean over K.
// Block = 128 thr; weights resident; 8 groups of 4 (b,g) per block,
// each wave owns one (b,g): 16 neighbor rows x 64 cols.
// ---------------------------------------------------------------------------
#define NM_GROUPS 8
__global__ void neighbor_msg_kernel(const float* __restrict__ pos,
                                    const __bf16* __restrict__ hbf,
                                    const int* __restrict__ idx,
                                    const float* __restrict__ peW1,
                                    const float* __restrict__ peB1,
                                    const float* __restrict__ peW2,
                                    const float* __restrict__ peB2,
                                    const float* __restrict__ msgW,
                                    const float* __restrict__ msgB,
                                    float* __restrict__ aggOut) {
  __shared__ __align__(16) __bf16 sPW1T[64][40];  // [n][k<=32]
  __shared__ __align__(16) __bf16 sPW2T[64][80];
  __shared__ __align__(16) __bf16 sMWT[64][80];
  __shared__ float sPB1[64], sPB2[64], sMB[64];
  __shared__ __align__(16) __bf16 sRel[4][16][40];
  __shared__ __align__(16) __bf16 sHj[4][16][64];
  __shared__ __align__(16) __bf16 sMid[4][16][80];
  __shared__ __align__(16) __bf16 sMsgIn[4][16][80];

  int tid = threadIdx.x;
  int wave = tid >> 5, lane = tid & 31;
  int half = lane >> 4, n16 = lane & 15;

  // stage shared weights (transposed) once per block
  for (int t = tid; t < 32 * 64; t += 128) {
    int k = t >> 6, n = t & 63;
    sPW1T[n][k] = (k < CIN) ? (__bf16)peW1[k * 64 + n] : (__bf16)0.0f;
  }
  for (int t = tid; t < 64 * 64; t += 128) {
    int k = t >> 6, n = t & 63;
    sPW2T[n][k] = (__bf16)peW2[k * 64 + n];
    sMWT[n][k]  = (__bf16)msgW[k * 64 + n];
  }
  if (tid < 64) { sPB1[tid] = peB1[tid]; sPB2[tid] = peB2[tid]; sMB[tid] = msgB[tid]; }

  for (int it = 0; it < NM_GROUPS; ++it) {
    int pair = (blockIdx.x * NM_GROUPS + it) * 4 + wave;   // (b,g)
    int b = pair / GG, g = pair - b * GG;
    float gx = vox_coord(g >> 8);
    float gy = vox_coord((g >> 4) & 15);
    float gz = vox_coord(g & 15);

    // per-wave staging of rel (k padded to 32 with zeros) and hj
#pragma unroll
    for (int e = 0; e < 16; ++e)
      sRel[wave][lane & 15][(lane >> 4) * 16 + e] = (__bf16)0.0f;
    if (lane < 16) {
      int nb = idx[((size_t)b * GG + g) * KK + lane];
      const float* pp = pos + ((size_t)b * NN + nb) * 3;
      sRel[wave][lane][0] = (__bf16)(pp[0] * 2.0f - 1.0f - gx);
      sRel[wave][lane][1] = (__bf16)(pp[1] * 2.0f - 1.0f - gy);
      sRel[wave][lane][2] = (__bf16)(pp[2] * 2.0f - 1.0f - gz);
      const uint4* hrow = (const uint4*)(hbf + ((size_t)b * NN + nb) * 64);
      uint4* dst = (uint4*)(&sHj[wave][lane][0]);
#pragma unroll
      for (int q = 0; q < 8; ++q) dst[q] = hrow[q];
    }
    __syncthreads();

    // pe layer 1: [16,32pad] @ [32pad,64]
    {
      v16bf a = load_A(&sRel[wave][0][0], 40);
#pragma unroll
      for (int nt = 0; nt < 4; ++nt) {
        v16bf bfrag = load_B_T(&sPW1T[nt * 16][0], 40);
        v8f acc = {};
        acc = wmma_bf16(a, bfrag, acc);
#pragma unroll
        for (int j = 0; j < 8; ++j) {
          int m = j + half * 8, col = nt * 16 + n16;
          sMid[wave][m][col] = (__bf16)gelu_t(acc[j] + sPB1[col]);
        }
      }
    }
    __syncthreads();

    // pe layer 2 + elementwise * hj
    {
      v16bf a0 = load_A(&sMid[wave][0][0], 80);
      v16bf a1 = load_A(&sMid[wave][0][32], 80);
#pragma unroll
      for (int nt = 0; nt < 4; ++nt) {
        v8f acc = {};
        acc = wmma_bf16(a0, load_B_T(&sPW2T[nt * 16][0], 80), acc);
        acc = wmma_bf16(a1, load_B_T(&sPW2T[nt * 16][32], 80), acc);
#pragma unroll
        for (int j = 0; j < 8; ++j) {
          int m = j + half * 8, col = nt * 16 + n16;
          float pe = acc[j] + sPB2[col];
          float hj = (float)sHj[wave][m][col];
          sMsgIn[wave][m][col] = (__bf16)(pe * hj);
        }
      }
    }
    __syncthreads();

    // msg GEMM + mean over K rows
    {
      v16bf a0 = load_A(&sMsgIn[wave][0][0], 80);
      v16bf a1 = load_A(&sMsgIn[wave][0][32], 80);
#pragma unroll
      for (int nt = 0; nt < 4; ++nt) {
        v8f acc = {};
        acc = wmma_bf16(a0, load_B_T(&sMWT[nt * 16][0], 80), acc);
        acc = wmma_bf16(a1, load_B_T(&sMWT[nt * 16][32], 80), acc);
        float part = 0.0f;
#pragma unroll
        for (int j = 0; j < 8; ++j) part += acc[j];
        part += __shfl_xor(part, 16, 32);     // combine row halves
        if (half == 0) {
          int col = nt * 16 + n16;
          aggOut[((size_t)b * GG + g) * 64 + col] = part * (1.0f / 16.0f) + sMB[col];
        }
      }
    }
    __syncthreads();
  }
}

// ---------------------------------------------------------------------------
// Kernel 5: conv3d 3x3x3 SAME (implicit GEMM over 27 taps) + bias
//           [+ residual] + gelu.  in/out/res bf16 NCDHW [B][64][16^3].
// Block = 128 thr; block = (b, y, 4-z column) of 16 x-voxels; wave = co-tile.
// Input staged channel-LAST (A frags contiguous), tap weights staged once
// per tap and reused for all 4 z rows (4 accumulators).
// ---------------------------------------------------------------------------
__global__ void conv3d_kernel(const __bf16* __restrict__ in,
                              const float* __restrict__ W,   // [co][ci][27]
                              const float* __restrict__ bias,
                              const __bf16* __restrict__ res, // may be null
                              __bf16* __restrict__ out) {
  __shared__ __align__(16) __bf16 sIn[6][3][18][80];  // zl, dy, x(-1..16), ci
  __shared__ __align__(16) __bf16 sWT[64][80];        // [co][ci] for one tap
  __shared__ float sBias[64];

  int tid = threadIdx.x;
  int wave = tid >> 5, lane = tid & 31;
  int half = lane >> 4, n16 = lane & 15;
  int nt = wave;
  int m = lane & 15;

  int b = blockIdx.x >> 6;
  int rem = blockIdx.x & 63;
  int y = rem >> 2;
  int z0 = (rem & 3) << 2;

  // stage shifted input planes with halo, zero-padded, channel-last
  for (int t = tid; t < 64 * 6 * 3 * 18; t += 128) {
    int ci = t / 324;
    int r = t - ci * 324;
    int zl = r / 54;
    int r2 = r - zl * 54;
    int dy = r2 / 18, xx = r2 - dy * 18;
    int zz = z0 + zl - 1, yy = y + dy - 1, x = xx - 1;
    float v = 0.0f;
    if (zz >= 0 && zz < 16 && yy >= 0 && yy < 16 && x >= 0 && x < 16)
      v = (float)in[((size_t)(b * 64 + ci) << 12) + (zz << 8) + (yy << 4) + x];
    sIn[zl][dy][xx][ci] = (__bf16)v;
  }
  if (tid < 64) sBias[tid] = bias[tid];
  __syncthreads();

  v8f acc[4] = {};
  for (int tap = 0; tap < 27; ++tap) {
    int kz = tap / 9, ky = (tap / 3) % 3, kx = tap % 3;
    // stage tap weights transposed: sWT[co][ci]
    for (int t = tid; t < 64 * 64; t += 128) {
      int co = t >> 6, ci = t & 63;
      sWT[co][ci] = (__bf16)W[(size_t)(co * 64 + ci) * 27 + tap];
    }
    // prefetch next tap's weights into cache (global_prefetch_b8)
    if (tap + 1 < 27) {
      int co = tid >> 6, ci = tid & 63;
      __builtin_prefetch(&W[(size_t)(co * 64 + ci) * 27 + tap + 1], 0, 1);
    }
    __syncthreads();

#pragma unroll
    for (int s = 0; s < 2; ++s) {
      v16bf bfrag = load_B_T(&sWT[nt * 16][s * 32], 80);
#pragma unroll
      for (int zi = 0; zi < 4; ++zi) {
        // A[mrow][k] = sIn[zi+kz][ky][mrow+kx][s*32 + k], k contiguous
        const __bf16* base = &sIn[zi + kz][ky][m + kx][s * 32 + half * 8];
        v16bf a;
#pragma unroll
        for (int e = 0; e < 8; ++e) a[e] = base[e];
#pragma unroll
        for (int e = 0; e < 8; ++e) a[8 + e] = base[16 + e];
        acc[zi] = wmma_bf16(a, bfrag, acc[zi]);
      }
    }
    __syncthreads();
  }

  int co = nt * 16 + n16;
#pragma unroll
  for (int zi = 0; zi < 4; ++zi) {
#pragma unroll
    for (int j = 0; j < 8; ++j) {
      int x = j + half * 8;
      size_t o = ((size_t)(b * 64 + co) << 12) + ((z0 + zi) << 8) + (y << 4) + x;
      float v = acc[zi][j] + sBias[co];
      if (res) v += (float)res[o];
      out[o] = (__bf16)gelu_t(v);
    }
  }
}

// ---------------------------------------------------------------------------
// Kernel 6a: global mean pool over G per (b,c).  Kernel 6b: classifier.
// ---------------------------------------------------------------------------
__global__ void pool_kernel(const __bf16* __restrict__ x,
                            float* __restrict__ pooled) {
  __shared__ float red[256];
  int bc = blockIdx.x;         // b*64 + c
  int tid = threadIdx.x;
  float s = 0.0f;
  for (int g = tid; g < GG; g += 256) s += (float)x[(size_t)bc * GG + g];
  red[tid] = s;
  __syncthreads();
  for (int w = 128; w > 0; w >>= 1) {
    if (tid < w) red[tid] += red[tid + w];
    __syncthreads();
  }
  if (tid == 0) pooled[bc] = red[0] * (1.0f / (float)GG);
}

__global__ void cls_kernel(const float* __restrict__ pooled,
                           const float* __restrict__ clsW,
                           const float* __restrict__ clsB,
                           float* __restrict__ outp) {
  int t = threadIdx.x;
  if (t < BB * COUT) {
    int b = t / COUT, o = t - b * COUT;
    float s = clsB[o];
    for (int c = 0; c < 64; ++c)
      s += pooled[b * 64 + c] * clsW[c * COUT + o];
    outp[t] = s;
  }
}

// ---------------------------------------------------------------------------
extern "C" void kernel_launch(void* const* d_in, const int* in_sizes, int n_in,
                              void* d_out, int out_size, void* d_ws, size_t ws_size,
                              hipStream_t stream) {
  const float* pos   = (const float*)d_in[0];
  const float* f     = (const float*)d_in[1];
  const float* ne_w1 = (const float*)d_in[2];
  const float* ne_b1 = (const float*)d_in[3];
  const float* ne_w2 = (const float*)d_in[4];
  const float* ne_b2 = (const float*)d_in[5];
  const float* pe_w1 = (const float*)d_in[6];
  const float* pe_b1 = (const float*)d_in[7];
  const float* pe_w2 = (const float*)d_in[8];
  const float* pe_b2 = (const float*)d_in[9];
  const float* msg_w = (const float*)d_in[10];
  const float* msg_b = (const float*)d_in[11];
  const float* up_w1 = (const float*)d_in[12];
  const float* up_b1 = (const float*)d_in[13];
  const float* up_w2 = (const float*)d_in[14];
  const float* up_b2 = (const float*)d_in[15];
  const float* cw1a  = (const float*)d_in[16];
  const float* cb1a  = (const float*)d_in[17];
  const float* cw1b  = (const float*)d_in[18];
  const float* cb1b  = (const float*)d_in[19];
  const float* cw2a  = (const float*)d_in[20];
  const float* cb2a  = (const float*)d_in[21];
  const float* cw2b  = (const float*)d_in[22];
  const float* cb2b  = (const float*)d_in[23];
  const float* cls_w = (const float*)d_in[24];
  const float* cls_b = (const float*)d_in[25];
  float* outp = (float*)d_out;

  char* ws = (char*)d_ws;
  __bf16* h_bf   = (__bf16*)(ws + 0);                 // [B*N,64] bf16  2MB
  int*    idxBuf = (int*)   (ws + 2097152);           // [B,G,16]       1MB
  float*  aggF   = (float*) (ws + 3145728);           // [B*G,64] f32   4MB
  __bf16* vox    = (__bf16*)(ws + 7340032);           // [B,64,G] bf16  2MB
  __bf16* buf1   = (__bf16*)(ws + 9437184);           // 2MB
  __bf16* buf2   = (__bf16*)(ws + 11534336);          // 2MB
  float*  pooled = (float*) (ws + 13631488);          // [B,64]

  // 1) node embedding h = mlp2(f)          [B*N,64] bf16
  mlp2_kernel<<<(BB * NN) / (16 * MLP_TILES), 128, 0, stream>>>(
      f, ne_w1, ne_b1, ne_w2, ne_b2, h_bf, BB * NN, CIN, 0);

  // 2) KNN: idx[B,G,16]
  knn_kernel<<<BB * (GG / 256), 256, 0, stream>>>(pos, idxBuf);

  // 3) fused gather + pe MLP + msg + mean-K  -> agg [B*G,64] f32
  neighbor_msg_kernel<<<(BB * GG) / (4 * NM_GROUPS), 128, 0, stream>>>(
      pos, h_bf, idxBuf, pe_w1, pe_b1, pe_w2, pe_b2, msg_w, msg_b, aggF);

  // 4) up MLP -> voxel features, stored NCDHW bf16 [B,64,G]
  mlp2_kernel<<<(BB * GG) / (16 * MLP_TILES), 128, 0, stream>>>(
      aggF, up_w1, up_b1, up_w2, up_b2, vox, BB * GG, HH, 1);

  // 5) two residual conv blocks (4-z column per block)
  conv3d_kernel<<<BB * 64, 128, 0, stream>>>(vox,  cw1a, cb1a, nullptr, buf1); // h1
  conv3d_kernel<<<BB * 64, 128, 0, stream>>>(buf1, cw1b, cb1b, vox,    buf2); // r1
  conv3d_kernel<<<BB * 64, 128, 0, stream>>>(buf2, cw2a, cb2a, nullptr, buf1); // h2
  conv3d_kernel<<<BB * 64, 128, 0, stream>>>(buf1, cw2b, cb2b, buf2,   vox);  // r2

  // 6) global mean pool + classifier
  pool_kernel<<<BB * 64, 256, 0, stream>>>(vox, pooled);
  cls_kernel<<<1, 256, 0, stream>>>(pooled, cls_w, cls_b, outp);
}
